// MemoryBank_75935021793842
// MI455X (gfx1250) — matrix operand order repaired and checked by reading, not compiled, not run on hardware
//
#include <hip/hip_runtime.h>

typedef __attribute__((ext_vector_type(16))) __bf16          v16bf;
typedef __attribute__((ext_vector_type(8)))  __bf16          v8bf;
typedef __attribute__((ext_vector_type(8)))  float           v8f;
typedef __attribute__((ext_vector_type(4)))  float           f4;
typedef __attribute__((ext_vector_type(4)))  unsigned short  v4u16;

#define MDIM 65536   // batch
#define KDIM 1024    // MEM_DIM
#define NDIM 1536    // 3 * NODE_FEAT_DIM
#define NF   512     // NODE_FEAT_DIM

__device__ __forceinline__ unsigned short f2bf(float f) {
  unsigned int u = __float_as_uint(f);
  u += 0x7FFFu + ((u >> 16) & 1u);   // round-to-nearest-even
  return (unsigned short)(u >> 16);
}

// One wave (32 lanes) per batch row: sum memory row, x = sum + embedding, bf16-convert.
__global__ void __launch_bounds__(256) prep_x_kernel(
    const int* __restrict__ ids, const float* __restrict__ mem,
    const float* __restrict__ emb, unsigned short* __restrict__ Xb) {
  const int row  = (blockIdx.x * 256 + threadIdx.x) >> 5;
  const int lane = threadIdx.x & 31;
  if (row >= MDIM) return;
  const int id = ids[row];
  const f4* mrow = (const f4*)(mem + (size_t)id * KDIM);
  const f4* erow = (const f4*)(emb + (size_t)id * KDIM);
  float s = 0.f;
#pragma unroll
  for (int i = 0; i < 8; ++i) {
    f4 m = __builtin_nontemporal_load(&mrow[i * 32 + lane]);
    s += (m.x + m.y) + (m.z + m.w);
  }
#pragma unroll
  for (int off = 16; off > 0; off >>= 1) s += __shfl_xor(s, off, 32);
  v4u16* xrow = (v4u16*)(Xb + (size_t)row * KDIM);
#pragma unroll
  for (int i = 0; i < 8; ++i) {
    f4 e = __builtin_nontemporal_load(&erow[i * 32 + lane]);
    v4u16 o;
    o.x = f2bf(s + e.x); o.y = f2bf(s + e.y);
    o.z = f2bf(s + e.z); o.w = f2bf(s + e.w);
    xrow[i * 32 + lane] = o;
  }
}

// Fuse+transpose Wq|Ws|Wk (each KDIM x NF, row-major f32) -> Wt (NDIM x KDIM, row-major bf16).
__global__ void __launch_bounds__(256) prep_w_kernel(
    const float* __restrict__ Wq, const float* __restrict__ Ws,
    const float* __restrict__ Wk, unsigned short* __restrict__ Wt) {
  const int ng  = blockIdx.x;          // global output column 0..NDIM-1
  const int seg = ng >> 9;
  const int n   = ng & (NF - 1);
  const float* W = (seg == 0) ? Wq : (seg == 1) ? Ws : Wk;
  unsigned short* dst = Wt + (size_t)ng * KDIM;
  for (int k = threadIdx.x; k < KDIM; k += 256)
    dst[k] = f2bf(W[(size_t)k * NF + n]);
}

// C = sigmoid(Xb @ Wt^T + bias).  Block tile 128x128 (4 waves, each 64x64 = 4x4 WMMA tiles).
__global__ void __launch_bounds__(128) gemm_wmma_kernel(
    const __bf16* __restrict__ Xb,   // MDIM x KDIM row-major
    const __bf16* __restrict__ Wt,   // NDIM x KDIM row-major (W transposed)
    const float* __restrict__ bq, const float* __restrict__ bs,
    const float* __restrict__ bk, float* __restrict__ out) {
  const int lane  = threadIdx.x & 31;
  const int wid   = threadIdx.x >> 5;                 // 0..3
  const int m_blk = blockIdx.y * 128 + (wid & 1) * 64;
  const int n_blk = blockIdx.x * 128 + (wid >> 1) * 64;
  const int lrow  = lane & 15;
  const int khalf = lane >> 4;                        // 0 or 1

  v8f acc[4][4];
#pragma unroll
  for (int i = 0; i < 4; ++i)
#pragma unroll
    for (int j = 0; j < 4; ++j) acc[i][j] = (v8f)(0.0f);

  for (int k0 = 0; k0 < KDIM; k0 += 32) {
    v16bf a[4], b[4];
    // A fragment (16x32 bf16): lanes 0-15 rows M, K = khalf*16..+15 contiguous
#pragma unroll
    for (int i = 0; i < 4; ++i) {
      const __bf16* ap = Xb + (size_t)(m_blk + i * 16 + lrow) * KDIM + k0 + khalf * 16;
      a[i] = *(const v16bf*)ap;
    }
    // B fragment (32x16 bf16): lane holds column n, K = khalf*16..+15 contiguous in Wt
#pragma unroll
    for (int j = 0; j < 4; ++j) {
      const __bf16* bp = Wt + (size_t)(n_blk + j * 16 + lrow) * KDIM + k0 + khalf * 16;
      b[j] = *(const v16bf*)bp;
    }
#pragma unroll
    for (int i = 0; i < 4; ++i)
#pragma unroll
      for (int j = 0; j < 4; ++j)
        acc[i][j] = __builtin_amdgcn_wmma_f32_16x16x32_bf16(
            false, a[i], false, b[j], (short)0, acc[i][j], false, false);
  }

  // Epilogue: bias + sigmoid; outputs q|s|k concatenated (seg = col>>9).
#pragma unroll
  for (int j = 0; j < 4; ++j) {
    const int col = n_blk + j * 16 + lrow;
    const int seg = col >> 9;
    const int c   = col & (NF - 1);
    const float* bias = (seg == 0) ? bq : (seg == 1) ? bs : bk;
    const float bv = bias[c];
    float* oseg = out + (size_t)seg * ((size_t)MDIM * NF);
#pragma unroll
    for (int i = 0; i < 4; ++i) {
#pragma unroll
      for (int r = 0; r < 8; ++r) {
        const int row = m_blk + i * 16 + r + ((lane >= 16) ? 8 : 0);
        float v = acc[i][j][r] + bv;
        v = 1.0f / (1.0f + __expf(-v));
        __builtin_nontemporal_store(v, &oseg[(size_t)row * NF + c]);
      }
    }
  }
}

extern "C" void kernel_launch(void* const* d_in, const int* in_sizes, int n_in,
                              void* d_out, int out_size, void* d_ws, size_t ws_size,
                              hipStream_t stream) {
  (void)in_sizes; (void)n_in; (void)out_size; (void)ws_size;
  const int*   ids = (const int*)d_in[0];
  const float* mem = (const float*)d_in[1];
  const float* emb = (const float*)d_in[2];
  const float* Wq  = (const float*)d_in[3];
  const float* bq  = (const float*)d_in[4];
  const float* Ws  = (const float*)d_in[5];
  const float* bs  = (const float*)d_in[6];
  const float* Wk  = (const float*)d_in[7];
  const float* bk  = (const float*)d_in[8];
  float* out = (float*)d_out;

  unsigned short* Xb = (unsigned short*)d_ws;                 // MDIM*KDIM bf16 (128 MB)
  unsigned short* Wt = Xb + (size_t)MDIM * KDIM;              // NDIM*KDIM bf16 (3 MB)

  prep_x_kernel<<<MDIM / 8, 256, 0, stream>>>(ids, mem, emb, Xb);
  prep_w_kernel<<<NDIM, 256, 0, stream>>>(Wq, Ws, Wk, Wt);
  gemm_wmma_kernel<<<dim3(NDIM / 128, MDIM / 128), 128, 0, stream>>>(
      (const __bf16*)Xb, (const __bf16*)Wt, bq, bs, bk, out);
}
